// LIFLayer_17059610100145
// MI455X (gfx1250) — matrix operand order repaired and checked by reading, not compiled, not run on hardware
//
#include <hip/hip_runtime.h>
#include <math.h>

typedef __attribute__((ext_vector_type(16))) _Float16 v16h;
typedef __attribute__((ext_vector_type(2)))  _Float16 h2;
typedef __attribute__((ext_vector_type(8)))  float    v8f;

#define T_STEPS 256
#define B_SZ    128
#define CIN     512
#define COUT    512
#define HIDN    256
#define NBLK    32
#define NTHR    256
#define ALPHA_MEM 0.9512294245007140f   /* exp(-0.001/0.02) */

// ---------------- workspace layout (bytes) ----------------
constexpr size_t OFF_BAR   = 0;                      // grid-barrier counter
constexpr size_t OFF_H     = 1024;                   // h f16      : 128*256*2 = 65536
constexpr size_t OFF_COMB  = 1024 + 65536;           // combined   : 128*512*2 = 131072
constexpr size_t OFF_XSLOW = 1024 + 65536 + 131072;  // [x,slow]   : 128*1024*2 = 262144

// ---------------- WMMA fragment helpers (ISA 7.12.2, wave32) ----------------
// A (16x32, f16): lanes 0-15 -> M=lane, K offsets {0..7,16..23};
//                 lanes 16-31 -> M=lane-16, K offsets {8..15,24..31}
__device__ __forceinline__ v16h load_a_frag(const _Float16* __restrict__ A,
                                            int lda, int m0, int k0, int lane) {
  const _Float16* p = A + (size_t)(m0 + (lane & 15)) * lda + k0 + ((lane & 16) ? 8 : 0);
  v16h a;
#pragma unroll
  for (int j = 0; j < 8; ++j) {
    int k = ((j & 4) ? 16 : 0) + (j & 3) * 2;
    h2 pr = *(const h2*)(p + k);             // merges into global_load_b128 pairs
    a[2 * j]     = pr[0];
    a[2 * j + 1] = pr[1];
  }
  return a;
}

#define WMMA_F16(acc, a, b) \
  __builtin_amdgcn_wmma_f32_16x16x32_f16(false, (a), false, (b), (short)0, (acc), false, false)

// ---------------- grid-wide barrier (monotonic epoch counter) ----------------
__device__ __forceinline__ void grid_sync(unsigned* bar, unsigned target) {
  __threadfence();                           // release my global writes (global_wb DEV)
  __syncthreads();
  if (threadIdx.x == 0) {
    __hip_atomic_fetch_add(bar, 1u, __ATOMIC_RELEASE, __HIP_MEMORY_SCOPE_AGENT);
    while (__hip_atomic_load(bar, __ATOMIC_ACQUIRE, __HIP_MEMORY_SCOPE_AGENT) < target)
      __builtin_amdgcn_s_sleep(2);
  }
  __syncthreads();
  __threadfence();                           // acquire: invalidate stale near caches
}

__global__ void reset_bar(unsigned* bar) {
  if (threadIdx.x == 0 && blockIdx.x == 0) *bar = 0u;
}

// ---------------- persistent LIF scan kernel ----------------
__global__ __launch_bounds__(NTHR) void lif_persistent(
    const float* __restrict__ spikes, const float* __restrict__ W,
    const float* __restrict__ w1, const float* __restrict__ b1,
    const float* __restrict__ w2, const float* __restrict__ b2,
    const float* __restrict__ p_af, const float* __restrict__ p_as,
    const float* __restrict__ p_gf, const float* __restrict__ p_gs,
    float* __restrict__ out, unsigned* __restrict__ bar,
    _Float16* __restrict__ hbuf, _Float16* __restrict__ comb,
    _Float16* __restrict__ xslow) {
  // per-WG weight slices in WMMA B-fragment order, resident for all 256 steps
  __shared__ __align__(32) _Float16 lds_w1[32 * 512];  // 32 KB : w1 col-tile nt1
  __shared__ __align__(32) _Float16 lds_w2[8 * 512];   //  8 KB : w2 col-tile bid
  __shared__ __align__(32) _Float16 lds_W[16 * 512];   // 16 KB : W  col-tile bid

  const int tid  = threadIdx.x;
  const int lane = tid & 31;
  const int wave = tid >> 5;
  const int bid  = blockIdx.x;

  // ---- pack weight slices f32->f16 into LDS (fragment order) ----
  const int nt1 = bid >> 1;                  // phase-1 N tile owned by this WG
  for (int i = tid; i < 32 * 512; i += NTHR) {
    int h = i & 15, ln = (i >> 4) & 31, kt = i >> 9;
    int k = (kt << 5) + ((ln & 16) ? 8 : 0) + (h & 7) + ((h & 8) ? 16 : 0);
    int n = (nt1 << 4) + (ln & 15);
    lds_w1[i] = (_Float16)w1[(size_t)k * HIDN + n];
  }
  for (int i = tid; i < 8 * 512; i += NTHR) {
    int h = i & 15, ln = (i >> 4) & 31, kt = i >> 9;
    int k = (kt << 5) + ((ln & 16) ? 8 : 0) + (h & 7) + ((h & 8) ? 16 : 0);
    int n = (bid << 4) + (ln & 15);
    lds_w2[i] = (_Float16)w2[(size_t)k * CIN + n];
  }
  for (int i = tid; i < 16 * 512; i += NTHR) {
    int h = i & 15, ln = (i >> 4) & 31, kt = i >> 9;
    int k = (kt << 5) + ((ln & 16) ? 8 : 0) + (h & 7) + ((h & 8) ? 16 : 0);
    int n = (bid << 4) + (ln & 15);
    lds_W[i] = (_Float16)W[(size_t)k * COUT + n];
  }
  __syncthreads();

  // ---- build [x_0, slow_0=0] controller input ----
  for (int i = bid * NTHR + tid; i < B_SZ * 2 * CIN; i += NBLK * NTHR) {
    int row = i >> 10, k = i & 1023;
    xslow[i] = (_Float16)((k < CIN) ? spikes[(size_t)row * CIN + k] : 0.f);
  }

  const float af = fminf(fmaxf(p_af[0], 0.f), 0.9999f);
  const float as = fminf(fmaxf(p_as[0], 0.f), 0.999999f);
  const float log_as = __logf(as + 1e-8f);
  const float gf = p_gf[0], gs = p_gs[0];

  // fixed (row,col) ownership -> recurrent state lives in registers
  const int colT   = (bid << 4) + (lane & 15);           // phase-2/3 column
  const int rbaseT = (wave << 4) + ((lane & 16) ? 8 : 0);
  const int col1   = (nt1 << 4) + (lane & 15);           // phase-1 column
  const float bias1 = b1[col1];
  const float bias2 = b2[colT];

  float fast_r[8], slow_r[8], v_r[8], rate_r[8];
#pragma unroll
  for (int r = 0; r < 8; ++r) { fast_r[r] = 0.f; slow_r[r] = 0.f; v_r[r] = 0.f; rate_r[r] = 0.f; }

  unsigned epoch = 0;
  grid_sync(bar, ++epoch * NBLK);            // xslow(t=0) visible everywhere

  for (int t = 0; t < T_STEPS; ++t) {
    // ---- phase 1: h = relu([x,slow] @ w1 + b1), tiles: mt=(bid&1)*4+wave, nt=nt1
    if (wave < 4) {
      int m0 = (((bid & 1) << 2) + wave) << 4;
      v8f acc = {};
#pragma unroll 4
      for (int kt = 0; kt < 32; ++kt) {      // K = 1024
        v16h a = load_a_frag(xslow, 2 * CIN, m0, kt * 32, lane);
        v16h b = *(const v16h*)(&lds_w1[(kt << 9) + (lane << 4)]);
        acc = WMMA_F16(acc, a, b);
      }
      int rbase = m0 + ((lane & 16) ? 8 : 0);
#pragma unroll
      for (int r = 0; r < 8; ++r) {
        float hv = acc[r] + bias1;
        hbuf[(size_t)(rbase + r) * HIDN + col1] = (_Float16)(hv > 0.f ? hv : 0.f);
      }
    }
    grid_sync(bar, ++epoch * NBLK);

    // ---- phase 2: logits = h @ w2 + b2; warp; trace update (regs); emit comb & next [x,slow]
    {
      int m0 = wave << 4;
      v8f acc = {};
#pragma unroll
      for (int kt = 0; kt < 8; ++kt) {       // K = 256
        v16h a = load_a_frag(hbuf, HIDN, m0, kt * 32, lane);
        v16h b = *(const v16h*)(&lds_w2[(kt << 9) + (lane << 4)]);
        acc = WMMA_F16(acc, a, b);
      }
      const float* x_t = spikes + (size_t)t * (B_SZ * CIN);
      const float* x_n = x_t + (B_SZ * CIN);
      bool has_next = (t + 1) < T_STEPS;
#pragma unroll
      for (int r = 0; r < 8; ++r) {
        int row = rbaseT + r;
        size_t idx = (size_t)row * CIN + colT;
        float logit = acc[r] + bias2;
        float warp = 1.0f / (1.0f + __expf(-logit));
        warp = fminf(fmaxf(warp, 0.05f), 0.95f);
        float alpha_ad = __expf(log_as * warp);
        float x = x_t[idx];
        fast_r[r] = af * fast_r[r] + x;
        slow_r[r] = alpha_ad * slow_r[r] + x;
        comb[idx] = (_Float16)(x + gf * fast_r[r] + gs * slow_r[r]);
        xslow[(size_t)row * (2 * CIN) + CIN + colT] = (_Float16)slow_r[r];
        xslow[(size_t)row * (2 * CIN) + colT] = (_Float16)(has_next ? x_n[idx] : 0.f);
      }
    }
    grid_sync(bar, ++epoch * NBLK);

    // ---- phase 3: cur = comb @ W; LIF membrane update (regs); spikes out
    {
      // prefetch next-next spike block into L2 (hides the only HBM read latency
      // on the serial chain; lowers to global_prefetch_b8)
      if (t + 2 < T_STEPS) {
        const float* x_p = spikes + (size_t)(t + 2) * (B_SZ * CIN);
#pragma unroll
        for (int r = 0; r < 8; ++r)
          __builtin_prefetch(x_p + (size_t)(rbaseT + r) * CIN + colT, 0, 1);
      }
      int m0 = wave << 4;
      v8f acc = {};
#pragma unroll 4
      for (int kt = 0; kt < 16; ++kt) {      // K = 512
        v16h a = load_a_frag(comb, CIN, m0, kt * 32, lane);
        v16h b = *(const v16h*)(&lds_W[(kt << 9) + (lane << 4)]);
        acc = WMMA_F16(acc, a, b);
      }
      float* out_t = out + (size_t)t * (B_SZ * COUT);
#pragma unroll
      for (int r = 0; r < 8; ++r) {
        size_t idx = (size_t)(rbaseT + r) * COUT + colT;
        float v = ALPHA_MEM * v_r[r] + (1.0f - ALPHA_MEM) * acc[r];
        float s = (v - 1.0f) > 0.f ? 1.f : 0.f;   // superspike(v - V_TH)
        v_r[r] = (s > 0.f) ? 0.f : v;             // reset
        rate_r[r] += s;
        __builtin_nontemporal_store(s, &out_t[idx]);  // write-once stream, keep L2 clean
      }
    }
    // no barrier: next iteration's post-phase-1 barrier orders comb WAR correctly
  }

  // ---- rates (register accumulators -> output tail) ----
  float* out_rates = out + (size_t)T_STEPS * (B_SZ * COUT);
#pragma unroll
  for (int r = 0; r < 8; ++r) {
    size_t idx = (size_t)(rbaseT + r) * COUT + colT;
    out_rates[idx] = rate_r[r] * (1.0f / (float)T_STEPS);
  }
}

extern "C" void kernel_launch(void* const* d_in, const int* in_sizes, int n_in,
                              void* d_out, int out_size, void* d_ws, size_t ws_size,
                              hipStream_t stream) {
  const float* spikes  = (const float*)d_in[0];
  const float* W       = (const float*)d_in[1];
  const float* p_af    = (const float*)d_in[2];
  const float* p_as    = (const float*)d_in[3];
  const float* p_gf    = (const float*)d_in[4];
  const float* p_gs    = (const float*)d_in[5];
  const float* ctrl_w1 = (const float*)d_in[6];
  const float* ctrl_b1 = (const float*)d_in[7];
  const float* ctrl_w2 = (const float*)d_in[8];
  const float* ctrl_b2 = (const float*)d_in[9];

  char* ws = (char*)d_ws;
  unsigned* bar   = (unsigned*)(ws + OFF_BAR);
  _Float16* hbuf  = (_Float16*)(ws + OFF_H);
  _Float16* comb  = (_Float16*)(ws + OFF_COMB);
  _Float16* xslow = (_Float16*)(ws + OFF_XSLOW);

  reset_bar<<<1, 32, 0, stream>>>(bar);
  lif_persistent<<<NBLK, NTHR, 0, stream>>>(
      spikes, W, ctrl_w1, ctrl_b1, ctrl_w2, ctrl_b2,
      p_af, p_as, p_gf, p_gs,
      (float*)d_out, bar, hbuf, comb, xslow);
}
